// Interaction_GraphConvolution_53206054863548
// MI455X (gfx1250) — compile-verified, matrix operand order checked by reading, not run on hardware
//
#include <hip/hip_runtime.h>

// N=512 nodes, F_IN=F_OUT=256.
// out[a,f] = sum_j P[a,f,j] * u[j,f] * sum_i A[a,i] * u[i,f] * H[i,f,j]
// Per f: C_f = A @ (diag(u_f) H_f)  (512x512x512 f16-WMMA GEMM, f32 accum),
// then out[:,f] = rowsum( C_f * P_f * u_f[j] ).
// 68.7 GFLOP of matmul vs ~540MB HBM traffic -> matrix-unit bound on MI455X.

#define NN 512
#define FD 256
#define TM 128
#define TN 128
#define TK 32
#define LDH 40   // padded LDS row stride in halves (80B, conflict-free b128)

typedef _Float16 v16h __attribute__((ext_vector_type(16)));
typedef _Float16 v8h  __attribute__((ext_vector_type(8)));
typedef _Float16 v4h  __attribute__((ext_vector_type(4)));
typedef float    v8f  __attribute__((ext_vector_type(8)));
typedef float    f32x4 __attribute__((ext_vector_type(4)));

// ---------------- zero the (atomic-accumulated) output --------------------
__global__ void zero_out_kernel(float* __restrict__ out) {
  out[blockIdx.x * blockDim.x + threadIdx.x] = 0.0f;
}

// ---------------- wfT[f][i] = (X @ W + b)[i, f] ---------------------------
// Transposed so each WG of the main kernel loads u_f = wfT[f,:] coalesced.
__global__ void wf_kernel(const float* __restrict__ nf,
                          const float* __restrict__ w,
                          const float* __restrict__ bias,
                          float* __restrict__ wfT) {
  const int i = blockIdx.x;    // node
  const int f = threadIdx.x;   // feature
  float s = bias[f];
  for (int k = 0; k < FD; ++k)
    s = fmaf(nf[i * FD + k], w[k * FD + f], s);   // nf broadcast, w coalesced
  wfT[f * NN + i] = s;
}

// ---------------- main WMMA kernel ----------------------------------------
__global__ __launch_bounds__(256)
void gconv_kernel(const float* __restrict__ A,    // [512,512] adjacency
                  const float* __restrict__ MH,   // [512,256,512] mask_hadamard
                  const float* __restrict__ MF,   // [512,256,512] mask_father
                  const float* __restrict__ wfT,  // [256,512]
                  float* __restrict__ out)        // [512,256]
{
  __shared__ __align__(16) _Float16 As[TM * LDH]; // A*u tile, [a][i] f16
  __shared__ __align__(16) _Float16 Bs[TN * LDH]; // H_f tile transposed [j][i]
  __shared__ float uS[NN];                        // u_f = wf[:, f]

  const int bx = blockIdx.x;          // 256 f * 4 aTile * 4 jTile
  const int f  = bx >> 4;             // f in high bits: 16 WGs/f adjacent -> H_f shared in L2
  const int a0 = ((bx >> 2) & 3) * TM;
  const int j0 = (bx & 3) * TN;

  const int tid  = threadIdx.x;
  const int wave = tid >> 5;
  const int lane = tid & 31;
  const int wm   = wave >> 1;         // 4 waves along M
  const int wn   = wave & 1;          // 2 waves along N
  const int l15  = lane & 15;
  const int hh   = lane >> 4;         // lane half selects K groups / M+8 rows

  for (int i = tid; i < NN; i += 256) uS[i] = wfT[f * NN + i];

  const v8f vz = {0.f,0.f,0.f,0.f,0.f,0.f,0.f,0.f};
  v8f acc[2][4];
  #pragma unroll
  for (int mi = 0; mi < 2; ++mi)
    #pragma unroll
    for (int ni = 0; ni < 4; ++ni) acc[mi][ni] = vz;

  for (int k0 = 0; k0 < NN; k0 += TK) {
    __syncthreads();   // also makes uS visible before first stage

    // Stage A' = A * diag(u): rows a0..a0+127, cols k0..k0+31 (f16)
    #pragma unroll
    for (int rr = 0; rr < 4; ++rr) {
      const int row = rr * 32 + (tid >> 3);
      const int c   = (tid & 7) * 4;
      f32x4 av = *reinterpret_cast<const f32x4*>(A + ((a0 + row) << 9) + k0 + c);
      v4h hv;
      hv[0] = (_Float16)(av[0] * uS[k0 + c + 0]);
      hv[1] = (_Float16)(av[1] * uS[k0 + c + 1]);
      hv[2] = (_Float16)(av[2] * uS[k0 + c + 2]);
      hv[3] = (_Float16)(av[3] * uS[k0 + c + 3]);
      *reinterpret_cast<v4h*>(As + row * LDH + c) = hv;   // 8B aligned
    }
    // Stage B^T: Bs[j][i] = f16(MH[k0+i, f, j0+j]); coalesced read, LDS transpose
    #pragma unroll
    for (int pp = 0; pp < 4; ++pp) {
      const int i = pp * 8 + (tid >> 5);
      const int j = lane * 4;
      f32x4 hv = *reinterpret_cast<const f32x4*>(
          MH + ((size_t)(k0 + i) << 17) + (f << 9) + j0 + j);
      Bs[(j + 0) * LDH + i] = (_Float16)hv[0];
      Bs[(j + 1) * LDH + i] = (_Float16)hv[1];
      Bs[(j + 2) * LDH + i] = (_Float16)hv[2];
      Bs[(j + 3) * LDH + i] = (_Float16)hv[3];
    }
    __syncthreads();

    // 16-bit A/B fragment layout: lanes 0-15 hold K{0..7,16..23},
    // lanes 16-31 hold K{8..15,24..31}; row = lane&15 within 16-tile.
    const int kh = hh * 8;
    v16h af[2], bf[4];
    #pragma unroll
    for (int mi = 0; mi < 2; ++mi) {
      const _Float16* p = As + (wm * 32 + mi * 16 + l15) * LDH + kh;
      v8h lo = *reinterpret_cast<const v8h*>(p);
      v8h hi = *reinterpret_cast<const v8h*>(p + 16);
      af[mi] = __builtin_shufflevector(lo, hi, 0,1,2,3,4,5,6,7,
                                                8,9,10,11,12,13,14,15);
    }
    #pragma unroll
    for (int ni = 0; ni < 4; ++ni) {
      const _Float16* p = Bs + (wn * 64 + ni * 16 + l15) * LDH + kh;
      v8h lo = *reinterpret_cast<const v8h*>(p);
      v8h hi = *reinterpret_cast<const v8h*>(p + 16);
      bf[ni] = __builtin_shufflevector(lo, hi, 0,1,2,3,4,5,6,7,
                                                8,9,10,11,12,13,14,15);
    }
    #pragma unroll
    for (int mi = 0; mi < 2; ++mi)
      #pragma unroll
      for (int ni = 0; ni < 4; ++ni)
        acc[mi][ni] = __builtin_amdgcn_wmma_f32_16x16x32_f16(
            false, af[mi], false, bf[ni], (short)0, acc[mi][ni], false, false);
  }

  // Epilogue: out[a,f] += sum_j C[a,j] * MF[a,f,j] * u[j]
  // C/D layout: VGPR r -> M = r + 8*hh, N = lane&15 (per 16x16 tile).
  #pragma unroll
  for (int mi = 0; mi < 2; ++mi) {
    const int abase = a0 + wm * 32 + mi * 16 + hh * 8;
    #pragma unroll
    for (int r = 0; r < 8; ++r) {
      const int a = abase + r;
      float s = 0.f;
      #pragma unroll
      for (int ni = 0; ni < 4; ++ni) {
        const int j = j0 + wn * 64 + ni * 16 + l15;
        const float p = MF[((size_t)a << 17) + (f << 9) + j];
        s += acc[mi][ni][r] * p * uS[j];
      }
      #pragma unroll
      for (int off = 1; off < 16; off <<= 1)   // reduce 16-lane j group
        s += __shfl_xor(s, off, 32);
      if (l15 == 0)
        atomicAdd(&out[a * FD + f], s);        // 8 contributors per (a,f)
    }
  }
}

extern "C" void kernel_launch(void* const* d_in, const int* in_sizes, int n_in,
                              void* d_out, int out_size, void* d_ws, size_t ws_size,
                              hipStream_t stream) {
  const float* nf   = (const float*)d_in[0];   // node_features   [512,256]
  const float* adj  = (const float*)d_in[1];   // adjacency       [512,512]
  const float* mfat = (const float*)d_in[2];   // mask_father     [512,256,512]
  /* d_in[3] = neighbor_count: unused by the reference math */
  const float* mhad = (const float*)d_in[4];   // mask_hadamard   [512,256,512]
  const float* w    = (const float*)d_in[5];   // weight          [256,256]
  const float* bias = (const float*)d_in[6];   // bias            [256]
  float* out = (float*)d_out;                  // [512,256]
  float* wfT = (float*)d_ws;                   // [256,512] = 512 KB scratch

  zero_out_kernel<<<(NN * FD) / 256, 256, 0, stream>>>(out);
  wf_kernel<<<NN, FD, 0, stream>>>(nf, w, bias, wfT);
  gconv_kernel<<<FD * 4 * 4, 256, 0, stream>>>(adj, mhad, mfat, wfT, out);
}
